// TimeAttention_61074434949402
// MI455X (gfx1250) — compile-verified, matrix operand order checked by reading, not run on hardware
//
#include <hip/hip_runtime.h>
#include <hip/hip_bf16.h>

// Problem constants (B,T,F,D,NH,HD,WIN) = (4,512,32,128,4,32,64)
#define TT   512
#define FF   32
#define DD   128
#define HID  512
#define MROWS 65536   // B*F*T

typedef __attribute__((ext_vector_type(16))) _Float16 v16h;
typedef __attribute__((ext_vector_type(8)))  _Float16 v8h;
typedef __attribute__((ext_vector_type(8)))  float    v8f;

__device__ __forceinline__ void lds_fence() {
  asm volatile("s_wait_dscnt 0" ::: "memory");
}

// Load a 16-half WMMA operand slice for one lane: two contiguous 128-bit loads.
// Canonical CDNA5 16-bit operand layout: lane.lo = row/col, lane.hi picks K
// groups {hi*8 .. hi*8+7} and {hi*8+16 .. hi*8+23}.
__device__ __forceinline__ v16h ld_ab(const _Float16* p) {
  v8h a = *(const v8h*)p;
  v8h b = *(const v8h*)(p + 16);
  v16h r;
#pragma unroll
  for (int i = 0; i < 8; ++i) { r[i] = a[i]; r[i + 8] = b[i]; }
  return r;
}

// ---------------------------------------------------------------------------
// f32 -> f16 weight conversion
// ---------------------------------------------------------------------------
__global__ void cvt_f16_kernel(const float* __restrict__ s, _Float16* __restrict__ d, int n) {
  int i = blockIdx.x * 256 + threadIdx.x;
  if (i < n) d[i] = (_Float16)s[i];
}

// ---------------------------------------------------------------------------
// Embed (outer product) + LayerNorm1 + RoPE.  One wave per (bf,t) row of 128.
// Writes hcf (f32, residual), h1 (f16, for V), qk (f16, for Q/K).
// ---------------------------------------------------------------------------
__global__ __launch_bounds__(256) void embed_ln_rope_kernel(
    const float* __restrict__ x, const float* __restrict__ W_in,
    const float* __restrict__ b_in, const float* __restrict__ g1,
    const float* __restrict__ be1, float* __restrict__ hcf,
    _Float16* __restrict__ h1, _Float16* __restrict__ qk) {
  int row  = (blockIdx.x * 256 + threadIdx.x) >> 5;  // bf*T + t
  int lane = threadIdx.x & 31;
  int bf = row >> 9, t = row & 511;
  int b  = bf >> 5, f = bf & 31;
  float xv = x[((size_t)b * TT + t) * FF + f];
  float v[4];
#pragma unroll
  for (int j = 0; j < 4; ++j) {
    int d = lane + 32 * j;
    v[j] = xv * W_in[d * FF + f] + b_in[d];
  }
  float* hp = hcf + (size_t)row * DD;
#pragma unroll
  for (int j = 0; j < 4; ++j) hp[lane + 32 * j] = v[j];
  // LayerNorm over D=128 (wave32 reduction)
  float s = v[0] + v[1] + v[2] + v[3];
#pragma unroll
  for (int o = 16; o >= 1; o >>= 1) s += __shfl_xor(s, o, 32);
  float mean = s * (1.0f / 128.0f);
  float q = 0.f;
#pragma unroll
  for (int j = 0; j < 4; ++j) { float d0 = v[j] - mean; q += d0 * d0; }
#pragma unroll
  for (int o = 16; o >= 1; o >>= 1) q += __shfl_xor(q, o, 32);
  float inv = rsqrtf(q * (1.0f / 128.0f) + 1e-5f);
  float hn[4];
  _Float16* h1p = h1 + (size_t)row * DD;
#pragma unroll
  for (int j = 0; j < 4; ++j) {
    int d = lane + 32 * j;
    hn[j] = (v[j] - mean) * inv * g1[d] + be1[d];
    h1p[d] = (_Float16)hn[j];
  }
  // RoPE: even/odd partner is d^1 -> adjacent lane
  _Float16* qkp = qk + (size_t)row * DD;
#pragma unroll
  for (int j = 0; j < 4; ++j) {
    int d = lane + 32 * j;
    float partner = __shfl_xor(hn[j], 1, 32);
    int k = (d & 31) >> 1;
    float ang = (float)t * __powf(10000.0f, -(float)k * (1.0f / 16.0f));
    float sn, cs;
    __sincosf(ang, &sn, &cs);
    float r = ((d & 1) == 0) ? (hn[j] * cs - partner * sn)
                             : (partner * sn + hn[j] * cs);
    qkp[d] = (_Float16)r;
  }
}

// ---------------------------------------------------------------------------
// LayerNorm2: f32 in -> f16 out, one wave per row.
// ---------------------------------------------------------------------------
__global__ __launch_bounds__(256) void ln_kernel(
    const float* __restrict__ h, const float* __restrict__ g,
    const float* __restrict__ be, _Float16* __restrict__ out) {
  int row = (blockIdx.x * 256 + threadIdx.x) >> 5;
  int lane = threadIdx.x & 31;
  const float* hp = h + (size_t)row * DD;
  float v[4];
#pragma unroll
  for (int j = 0; j < 4; ++j) v[j] = hp[lane + 32 * j];
  float s = v[0] + v[1] + v[2] + v[3];
#pragma unroll
  for (int o = 16; o >= 1; o >>= 1) s += __shfl_xor(s, o, 32);
  float mean = s * (1.0f / 128.0f);
  float q = 0.f;
#pragma unroll
  for (int j = 0; j < 4; ++j) { float d0 = v[j] - mean; q += d0 * d0; }
#pragma unroll
  for (int o = 16; o >= 1; o >>= 1) q += __shfl_xor(q, o, 32);
  float inv = rsqrtf(q * (1.0f / 128.0f) + 1e-5f);
  _Float16* op = out + (size_t)row * DD;
#pragma unroll
  for (int j = 0; j < 4; ++j) {
    int d = lane + 32 * j;
    op[d] = (_Float16)((v[j] - mean) * inv * g[d] + be[d]);
  }
}

// ---------------------------------------------------------------------------
// Generic WMMA GEMM: Y[M,N] = X[M,K] * W[N,K]^T + bias (+residual)(gelu).
// One wave per 16-row M tile; loops N/16 tiles x K/32 chunks of
// v_wmma_f32_16x16x32_f16.
// ---------------------------------------------------------------------------
template <bool OUT_F32, bool GELU, bool RESID, bool VT>
__global__ __launch_bounds__(256) void gemm_wmma_kernel(
    const _Float16* __restrict__ X, const _Float16* __restrict__ W,
    const float* __restrict__ bias, void* __restrict__ Y,
    const float* __restrict__ R, int M, int N, int K) {
  int wave = (blockIdx.x * 256 + threadIdx.x) >> 5;
  if (wave * 16 >= M) return;  // uniform per wave
  int lane = threadIdx.x & 31;
  int lo = lane & 15, hi = lane >> 4;
  int m_base = wave * 16;
  const int KC = K >> 5;
  const int NT = N >> 4;
  for (int nt = 0; nt < NT; ++nt) {
    int nb = nt << 4;
    v8f acc = {};
    const _Float16* xp = X + (size_t)(m_base + lo) * K + hi * 8;
    const _Float16* wp = W + (size_t)(nb + lo) * K + hi * 8;
    for (int kc = 0; kc < KC; ++kc) {
      v16h a = ld_ab(xp + kc * 32);
      v16h bv = ld_ab(wp + kc * 32);
      acc = __builtin_amdgcn_wmma_f32_16x16x32_f16(false, a, false, bv,
                                                   (short)0, acc, false, false);
    }
    float bn = bias[nb + lo];
#pragma unroll
    for (int r = 0; r < 8; ++r) {
      int m = m_base + r + hi * 8;
      int n = nb + lo;
      float v = acc[r] + bn;
      if (RESID) v += R[(size_t)m * N + n];
      if (GELU) v = 0.5f * v * (1.0f + erff(v * 0.70710678118654752f));
      if (OUT_F32) {
        ((float*)Y)[(size_t)m * N + n] = v;
      } else if (VT) {  // V stored transposed per-bf: Vt[bf][d][t]
        int bfi = m >> 9, t = m & 511;
        ((_Float16*)Y)[(((size_t)bfi) * DD + n) * TT + t] = (_Float16)v;
      } else {
        ((_Float16*)Y)[(size_t)m * N + n] = (_Float16)v;
      }
    }
  }
}

// ---------------------------------------------------------------------------
// Windowed causal attention (j<=i, i-j<=64). One wave per (bf, head, 16-query
// block). 3 chunks of 32 keys cover the 80-key window; online softmax.
// ---------------------------------------------------------------------------
__global__ __launch_bounds__(256) void attn_kernel(
    const _Float16* __restrict__ Q, const _Float16* __restrict__ Kb,
    const _Float16* __restrict__ Vt, _Float16* __restrict__ O) {
  __shared__ float    s_sc[8][16 * 32];
  __shared__ _Float16 s_p[8][16 * 32];
  __shared__ float    s_row[8][16];
  int wid  = (blockIdx.x * 256 + threadIdx.x) >> 5;
  int w    = (threadIdx.x >> 5) & 7;
  int lane = threadIdx.x & 31;
  int lo = lane & 15, hi = lane >> 4;
  int qb = wid & 31, head = (wid >> 5) & 3, bf = wid >> 7;
  const _Float16* Qp = Q + ((size_t)bf * TT) * DD + head * 32;
  const _Float16* Kp = Kb + ((size_t)bf * TT) * DD + head * 32;
  const _Float16* Vp = Vt + ((size_t)bf * DD + head * 32) * TT;
  int kstart = (qb >= 4) ? (qb - 4) * 16 : 0;
  v16h aq = ld_ab(Qp + (size_t)(qb * 16 + lo) * DD + hi * 8);  // whole HD=32 = one K-dim
  v8f o0 = {}, o1 = {};
  float m_run = -INFINITY, l_run = 0.0f;  // valid in lanes 0..15 (lane == row)
  const float rs = 0.17677669529663687f;  // 1/sqrt(32)
  for (int c = 0; c < 3; ++c) {
    int kb = kstart + 32 * c;
    // ---- scores: 16 queries x 32 keys via two WMMAs (K-dim = HD = 32) ----
#pragma unroll
    for (int nt = 0; nt < 2; ++nt) {
      int key = kb + nt * 16 + lo;
      int keyc = key < (TT - 1) ? key : (TT - 1);
      v16h bk = ld_ab(Kp + (size_t)keyc * DD + hi * 8);
      v8f sc = {};
      sc = __builtin_amdgcn_wmma_f32_16x16x32_f16(false, aq, false, bk,
                                                  (short)0, sc, false, false);
#pragma unroll
      for (int r = 0; r < 8; ++r) {
        int i = qb * 16 + r + hi * 8;
        int j = key;
        bool ok = (j <= i) && (i - j <= 64);
        s_sc[w][(r + hi * 8) * 32 + nt * 16 + lo] = ok ? sc[r] * rs : -INFINITY;
      }
    }
    lds_fence();
    // ---- online softmax: lane == row owns one query row ----
    if (lane < 16) {
      int rr = lane;
      float cmax = -INFINITY;
#pragma unroll
      for (int k2 = 0; k2 < 32; ++k2) cmax = fmaxf(cmax, s_sc[w][rr * 32 + k2]);
      float nm = fmaxf(m_run, cmax);
      float corr = __expf(m_run - nm);
      float ssum = 0.0f;
#pragma unroll
      for (int k2 = 0; k2 < 32; ++k2) {
        float p = __expf(s_sc[w][rr * 32 + k2] - nm);
        ssum += p;
        s_p[w][rr * 32 + k2] = (_Float16)p;
      }
      l_run = l_run * corr + ssum;
      m_run = nm;
      s_row[w][rr] = corr;
    }
    lds_fence();
    // rescale accumulated output by per-row correction
#pragma unroll
    for (int r = 0; r < 8; ++r) {
      float cr = s_row[w][r + hi * 8];
      o0[r] *= cr;
      o1[r] *= cr;
    }
    // P operand (A layout) straight out of LDS: two 128-bit LDS loads
    v16h ap;
    {
      const _Float16* pp = &s_p[w][lo * 32 + hi * 8];
      v8h x0 = *(const v8h*)pp;
      v8h x1 = *(const v8h*)(pp + 16);
#pragma unroll
      for (int i2 = 0; i2 < 8; ++i2) { ap[i2] = x0[i2]; ap[i2 + 8] = x1[i2]; }
    }
    // V operands from transposed V (contiguous along keys)
    v16h bv0 = ld_ab(Vp + (size_t)lo * TT + kb + hi * 8);
    v16h bv1 = ld_ab(Vp + (size_t)(16 + lo) * TT + kb + hi * 8);
    o0 = __builtin_amdgcn_wmma_f32_16x16x32_f16(false, ap, false, bv0, (short)0,
                                                o0, false, false);
    o1 = __builtin_amdgcn_wmma_f32_16x16x32_f16(false, ap, false, bv1, (short)0,
                                                o1, false, false);
    lds_fence();
  }
  if (lane < 16) s_row[w][lane] = 1.0f / l_run;
  lds_fence();
  _Float16* Op = O + ((size_t)bf * TT) * DD + head * 32;
#pragma unroll
  for (int r = 0; r < 8; ++r) {
    float il = s_row[w][r + hi * 8];
    size_t qrow = (size_t)(qb * 16 + r + hi * 8);
    Op[qrow * DD + lo]      = (_Float16)(o0[r] * il);
    Op[qrow * DD + 16 + lo] = (_Float16)(o1[r] * il);
  }
}

// ---------------------------------------------------------------------------
// Finalize: out = x + mean_d(h4) + (x @ W_in^T + b_in) @ W_out^T + b_out.
// One wave per (b,t) row (2048 rows total; tiny).
// ---------------------------------------------------------------------------
__global__ __launch_bounds__(256) void finalize_kernel(
    const float* __restrict__ x, const float* __restrict__ W_in,
    const float* __restrict__ b_in, const float* __restrict__ W_out,
    const float* __restrict__ b_out, const float* __restrict__ h4,
    float* __restrict__ out) {
  __shared__ float tmp[8][DD];
  int row = (blockIdx.x * 256 + threadIdx.x) >> 5;  // b*T + t
  int w = (threadIdx.x >> 5) & 7;
  int lane = threadIdx.x & 31;
  int b = row >> 9, t = row & 511;
  float xv = x[(size_t)row * FF + lane];  // lane indexes f
  float acc[4];
#pragma unroll
  for (int j = 0; j < 4; ++j) acc[j] = b_in[lane + 32 * j];
  for (int f2 = 0; f2 < 32; ++f2) {
    float xf = __shfl(xv, f2, 32);
#pragma unroll
    for (int j = 0; j < 4; ++j) acc[j] += xf * W_in[(lane + 32 * j) * FF + f2];
  }
#pragma unroll
  for (int j = 0; j < 4; ++j) tmp[w][lane + 32 * j] = acc[j];
  lds_fence();
  int f = lane;
  int bf = b * FF + f;
  const float* hrow = h4 + ((size_t)bf * TT + t) * DD;
  float ms = 0.0f;
  for (int d = 0; d < DD; ++d) ms += hrow[d];
  ms *= (1.0f / 128.0f);
  float yp = b_out[f];
  for (int d = 0; d < DD; ++d) yp += tmp[w][d] * W_out[f * DD + d];
  out[(size_t)row * FF + f] = xv + ms + yp;
}

// ---------------------------------------------------------------------------
// Host launcher
// ---------------------------------------------------------------------------
extern "C" void kernel_launch(void* const* d_in, const int* in_sizes, int n_in,
                              void* d_out, int out_size, void* d_ws,
                              size_t ws_size, hipStream_t stream) {
  (void)in_sizes; (void)n_in; (void)out_size; (void)ws_size;
  const float* x      = (const float*)d_in[0];
  const float* W_in   = (const float*)d_in[1];
  const float* b_in   = (const float*)d_in[2];
  const float* gamma1 = (const float*)d_in[3];
  const float* beta1  = (const float*)d_in[4];
  const float* gamma2 = (const float*)d_in[5];
  const float* beta2  = (const float*)d_in[6];
  const float* W_qkv  = (const float*)d_in[7];
  const float* b_qkv  = (const float*)d_in[8];
  const float* W_o    = (const float*)d_in[9];
  const float* b_o    = (const float*)d_in[10];
  const float* W1     = (const float*)d_in[11];
  const float* b1     = (const float*)d_in[12];
  const float* W2     = (const float*)d_in[13];
  const float* b2     = (const float*)d_in[14];
  const float* W_out  = (const float*)d_in[15];
  const float* b_out  = (const float*)d_in[16];

  char* ws = (char*)d_ws;
  const size_t M = MROWS;
  // Workspace layout (byte offsets), with reuse; everything stays L2-resident.
  float*    h_f32 = (float*)(ws);                            // 32MB: hcf->h2->h4
  _Float16* qk    = (_Float16*)(ws + (((size_t)32) << 20));  // 16MB (reused as h3)
  _Float16* h1    = (_Float16*)(ws + (((size_t)48) << 20));  // 16MB
  _Float16* Qb    = (_Float16*)(ws + (((size_t)64) << 20));  // 16MB
  _Float16* Kbuf  = (_Float16*)(ws + (((size_t)80) << 20));  // 16MB
  _Float16* Vt    = (_Float16*)(ws + (((size_t)96) << 20));  // 16MB
  _Float16* Obuf  = (_Float16*)(ws + (((size_t)112) << 20)); // 16MB
  _Float16* h3    = qk;                                      // reuse (qk dead)
  _Float16* G     = (_Float16*)(ws + (((size_t)64) << 20));  // 64MB over Q/K/Vt/O
  char* wtp = ws + (((size_t)128) << 20) + 4096;
  _Float16* Wq_h = (_Float16*)wtp;            wtp += 16384 * 2;
  _Float16* Wk_h = (_Float16*)wtp;            wtp += 16384 * 2;
  _Float16* Wv_h = (_Float16*)wtp;            wtp += 16384 * 2;
  _Float16* Wo_h = (_Float16*)wtp;            wtp += 16384 * 2;
  _Float16* W1_h = (_Float16*)wtp;            wtp += 65536 * 2;
  _Float16* W2_h = (_Float16*)wtp;            wtp += 65536 * 2;

  // Convert weights to f16.
  cvt_f16_kernel<<<64, 256, 0, stream>>>(W_qkv,          Wq_h, 16384);
  cvt_f16_kernel<<<64, 256, 0, stream>>>(W_qkv + 16384,  Wk_h, 16384);
  cvt_f16_kernel<<<64, 256, 0, stream>>>(W_qkv + 32768,  Wv_h, 16384);
  cvt_f16_kernel<<<64, 256, 0, stream>>>(W_o,            Wo_h, 16384);
  cvt_f16_kernel<<<256, 256, 0, stream>>>(W1,            W1_h, 65536);
  cvt_f16_kernel<<<256, 256, 0, stream>>>(W2,            W2_h, 65536);

  // 1) embed + LN1 + RoPE
  embed_ln_rope_kernel<<<8192, 256, 0, stream>>>(x, W_in, b_in, gamma1, beta1,
                                                 h_f32, h1, qk);
  // 2) Q/K/V projections (V stored transposed per bf)
  const int gemm_blocks = (int)(M / 16 * 32 / 256);  // 512
  gemm_wmma_kernel<false, false, false, false><<<gemm_blocks, 256, 0, stream>>>(
      qk, Wq_h, b_qkv, Qb, nullptr, (int)M, DD, DD);
  gemm_wmma_kernel<false, false, false, false><<<gemm_blocks, 256, 0, stream>>>(
      qk, Wk_h, b_qkv + DD, Kbuf, nullptr, (int)M, DD, DD);
  gemm_wmma_kernel<false, false, false, true><<<gemm_blocks, 256, 0, stream>>>(
      h1, Wv_h, b_qkv + 2 * DD, Vt, nullptr, (int)M, DD, DD);
  // 3) windowed attention
  attn_kernel<<<2048, 256, 0, stream>>>(Qb, Kbuf, Vt, Obuf);
  // 4) output projection + residual (h2 = hcf + y, in place)
  gemm_wmma_kernel<true, false, true, false><<<gemm_blocks, 256, 0, stream>>>(
      Obuf, Wo_h, b_o, h_f32, h_f32, (int)M, DD, DD);
  // 5) LN2
  ln_kernel<<<8192, 256, 0, stream>>>(h_f32, gamma2, beta2, h3);
  // 6) FFN up + exact gelu
  gemm_wmma_kernel<false, true, false, false><<<gemm_blocks, 256, 0, stream>>>(
      h3, W1_h, b1, G, nullptr, (int)M, HID, DD);
  // 7) FFN down + residual (h4 = h2 + ffn, in place)
  gemm_wmma_kernel<true, false, true, false><<<gemm_blocks, 256, 0, stream>>>(
      G, W2_h, b2, h_f32, h_f32, (int)M, DD, HID);
  // 8) mean over D + tiny y_proj + final sum
  finalize_kernel<<<256, 256, 0, stream>>>(x, W_in, b_in, W_out, b_out, h_f32,
                                           (float*)d_out);
}
